// HiggsAudioDualFFNFastDecoderLayer_70566312673880
// MI455X (gfx1250) — matrix operand order repaired and checked by reading, not compiled, not run on hardware
//
#include <hip/hip_runtime.h>
#include <stdint.h>

// ---------------------------------------------------------------------------
// HiggsAudio dual-FFN decoder layer for gfx1250 (MI455X).
// bf16 WMMA GEMMs fed by TDM (tensor_load_to_lds) double-buffered LDS tiles.
// B=2 S=2048 D=2048 H=16 HD=128 F=8192
// ---------------------------------------------------------------------------

typedef __bf16 bf16;
typedef __bf16 v16bf __attribute__((ext_vector_type(16)));
typedef float  v8f   __attribute__((ext_vector_type(8)));
typedef unsigned int u32x4 __attribute__((ext_vector_type(4)));
typedef int i32x4 __attribute__((ext_vector_type(4)));
typedef int i32x8 __attribute__((ext_vector_type(8)));

#define B_   2
#define S_   2048
#define D_   2048
#define H_   16
#define HD_  128
#define F_   8192
#define ROWS (B_ * S_)           // 4096
#define EPSV 1e-5f

union FragU { uint4 q[2]; v16bf v; };

// A-matrix 16x32 bf16 fragment: lane half selects K in {kb..kb+7, 16+kb..16+kb+7}
__device__ inline v16bf frag_a(const bf16* rowp, int kb) {
  FragU u;
  u.q[0] = *(const uint4*)(rowp + kb);
  u.q[1] = *(const uint4*)(rowp + 16 + kb);
  return u.v;
}
// B-matrix 32x16 bf16 fragment: lane half selects K in {h16/2..h16/2+7, 16+h16/2..16+h16/2+7}
__device__ inline v16bf frag_b(const bf16* rowp, int h16) {
  FragU u;
  u.q[0] = *(const uint4*)(rowp + (h16 >> 1));
  u.q[1] = *(const uint4*)(rowp + 16 + (h16 >> 1));
  return u.v;
}
__device__ inline v8f wmma_bf16(v16bf a, v16bf b, v8f c) {
  return __builtin_amdgcn_wmma_f32_16x16x32_bf16(false, a, false, b, (short)0,
                                                 c, false, false);
}

// ---------------------------------------------------------------------------
// TDM: DMA one 128-row x 32-col bf16 tile (row stride = rowlen elems) from
// global memory into LDS, with hardware padding of 4 DWORDs after every 16
// DWORDs so the LDS tile has a 40-element row stride (bank-conflict free,
// matches frag_a/frag_b addressing).
// D# packing per CDNA5 ISA. Issued by one wave per WG;
// completion tracked with TENSORcnt.
// ---------------------------------------------------------------------------
__device__ inline void tdm_load_tile(const bf16* gaddr, bf16* lds,
                                     int rowlen, int nrows)
{
  const unsigned ldsOff = (unsigned)(uintptr_t)lds;          // LDS byte offset
  const unsigned long long ga = (unsigned long long)(uintptr_t)gaddr;

  u32x4 g0;
  g0[0] = 1u;                                  // count=1, user mode, no gather
  g0[1] = ldsOff;                              // lds_addr
  g0[2] = (unsigned)(ga & 0xFFFFFFFFu);        // global_addr[31:0]
  g0[3] = (unsigned)((ga >> 32) & 0x1FFFFFFu)  // global_addr[56:32]
          | (2u << 30);                        // type=2 ("image")

  i32x8 g1;
  g1[0] = (1 << 16)      // data_size = 1 -> 2 bytes
        | (1 << 20)      // pad_enable
        | (3 << 22)      // pad_interval = 3 -> every 16 DWORDs (64B row)
        | (3 << 25);     // pad_amount   = 3 -> 4 DWORDs (16B) padding
  g1[1] = (rowlen & 0xFFFF) << 16;                       // tensor_dim0[15:0]
  g1[2] = ((unsigned)rowlen >> 16) | ((nrows & 0xFFFF) << 16); // td0 hi, td1 lo
  g1[3] = ((unsigned)nrows >> 16) | (32 << 16);          // td1 hi, tile_dim0=32
  g1[4] = 128;                                           // tile_dim1=128
  g1[5] = rowlen;                                        // dim0_stride[31:0]
  g1[6] = 0;                                             // stride hi, d1 stride
  g1[7] = 0;

  i32x4 gz = {0, 0, 0, 0};
#if __clang_major__ >= 23
  i32x8 gz8 = {0, 0, 0, 0, 0, 0, 0, 0};
  __builtin_amdgcn_tensor_load_to_lds(g0, g1, gz, gz, gz8, 0);
#else
  __builtin_amdgcn_tensor_load_to_lds(g0, g1, gz, gz, 0);
#endif
}

// ---------------------------------------------------------------------------
// Weight prep: fp32 [K][N] -> bf16 [N][K]  (transpose + convert, one pass)
// ---------------------------------------------------------------------------
__global__ __launch_bounds__(256)
void transpose_cvt_kernel(const float* __restrict__ W, bf16* __restrict__ Wt,
                          int K, int N)
{
  __shared__ float t[32][33];
  const int nb = blockIdx.x * 32;
  const int kbk = blockIdx.y * 32;
  const int tx = threadIdx.x & 31;
  const int ty = threadIdx.x >> 5;   // 0..7
  for (int j = ty; j < 32; j += 8)
    t[j][tx] = W[(size_t)(kbk + j) * N + nb + tx];
  __syncthreads();
  for (int j = ty; j < 32; j += 8)
    Wt[(size_t)(nb + j) * K + kbk + tx] = (bf16)t[tx][j];
}

// ---------------------------------------------------------------------------
// RMSNorm: fp32 row [D_] -> bf16 row
// ---------------------------------------------------------------------------
__device__ inline float wave_sum(float v) {
  for (int o = 16; o > 0; o >>= 1) v += __shfl_xor(v, o, 32);
  return v;
}

__global__ __launch_bounds__(256)
void rmsnorm_kernel(const float* __restrict__ x, const float* __restrict__ w,
                    bf16* __restrict__ out)
{
  const int row = blockIdx.x;
  const float* xr = x + (size_t)row * D_;
  const int tid = threadIdx.x;
  float ss = 0.0f;
  for (int i = tid; i < D_; i += 256) { const float v = xr[i]; ss += v * v; }
  ss = wave_sum(ss);
  __shared__ float red[8];
  if ((tid & 31) == 0) red[tid >> 5] = ss;
  __syncthreads();
  float tot = 0.0f;
  for (int i = 0; i < 8; ++i) tot += red[i];
  const float rsc = rsqrtf(tot * (1.0f / D_) + EPSV);
  bf16* orow = out + (size_t)row * D_;
  for (int i = tid; i < D_; i += 256) orow[i] = (bf16)(xr[i] * rsc * w[i]);
}

// ---------------------------------------------------------------------------
// Main bf16 WMMA GEMM: C[M,N] = A[M,K] * Bt[N,K]^T
// mode 0: C = result (fp32 overwrite)
// mode 1: C += result on rows where (mask[row]!=0)==(target!=0)
// Block: 256 threads (8 waves), tile 128x128, K-step 32.
// LDS tiles are DMA'd by the Tensor Data Mover (wave 0 issues descriptors),
// double-buffered, synchronized with s_wait_tensorcnt + workgroup barriers.
// ---------------------------------------------------------------------------
__global__ __launch_bounds__(256)
void gemm_bf16_wmma(const bf16* __restrict__ A, const bf16* __restrict__ Bt,
                    float* __restrict__ C, int M, int N, int K,
                    int mode, const unsigned char* __restrict__ mask, int target)
{
  __shared__ bf16 As[2][128 * 40];   // 128 rows x 32 K, stride 40 (TDM pad)
  __shared__ bf16 Bs[2][128 * 40];
  const int tid  = threadIdx.x;
  const int lane = tid & 31;
  const int wave = tid >> 5;
  const int wm   = (wave & 1) * 64;
  const int wn   = (wave >> 1) * 32;
  const int m0   = blockIdx.y * 128;
  const int n0   = blockIdx.x * 128;
  const int half = lane >> 4;
  const int l16  = lane & 15;
  const int kb   = half * 8;
  const int h16  = half * 16;
  const int nT   = K >> 5;           // number of 32-wide K tiles

  const bf16* Abase = A  + (size_t)m0 * K;
  const bf16* Bbase = Bt + (size_t)n0 * K;

  // Prologue: wave 0 fills both LDS buffers via TDM.
  if (wave == 0) {
    tdm_load_tile(Abase, &As[0][0], K, M - m0);
    tdm_load_tile(Bbase, &Bs[0][0], K, N - n0);
    if (nT > 1) {
      tdm_load_tile(Abase + 32, &As[1][0], K, M - m0);
      tdm_load_tile(Bbase + 32, &Bs[1][0], K, N - n0);
    }
  }

  v8f acc[4][2];
  for (int i = 0; i < 4; ++i)
    for (int j = 0; j < 2; ++j)
      for (int r = 0; r < 8; ++r) acc[i][j][r] = 0.0f;

  for (int it = 0; it < nT; ++it) {
    const int cur = it & 1;
    if (wave == 0) {
      // TDM ops from one wave complete in order: <=2 outstanding means the
      // current tile's A+B pair has landed (the 2 remaining are tile it+1).
      if (it + 1 < nT) __builtin_amdgcn_s_wait_tensorcnt(2);
      else             __builtin_amdgcn_s_wait_tensorcnt(0);
    }
    __syncthreads();                 // publish LDS tile to all waves

    const bf16* as = &As[cur][0];
    const bf16* bs = &Bs[cur][0];
    const v16bf bf0 = frag_b(&bs[(wn + l16) * 40], h16);
    const v16bf bf1 = frag_b(&bs[(wn + 16 + l16) * 40], h16);
    for (int i = 0; i < 4; ++i) {
      const v16bf af = frag_a(&as[(wm + i * 16 + l16) * 40], kb);
      acc[i][0] = wmma_bf16(af, bf0, acc[i][0]);
      acc[i][1] = wmma_bf16(af, bf1, acc[i][1]);
    }

    __syncthreads();                 // all waves done reading buffer `cur`
    if (wave == 0 && it + 2 < nT) {  // refill consumed buffer two tiles ahead
      const int kn = (it + 2) * 32;
      tdm_load_tile(Abase + kn, &As[cur][0], K, M - m0);
      tdm_load_tile(Bbase + kn, &Bs[cur][0], K, N - n0);
    }
  }

  // Epilogue. C layout: VGPR r -> row (r + 8*half), col = l16 within subtile.
  for (int i = 0; i < 4; ++i) {
    const int mBase = m0 + wm + i * 16 + 8 * half;
    for (int j = 0; j < 2; ++j) {
      const int nIdx = n0 + wn + j * 16 + l16;
      size_t idx = (size_t)mBase * N + nIdx;
      for (int r = 0; r < 8; ++r, idx += (size_t)N) {
        const float v = acc[i][j][r];
        if (mode == 0) {
          C[idx] = v;
        } else {
          const int row = mBase + r;
          if ((mask[row] != 0) == (target != 0)) C[idx] += v;
        }
      }
    }
  }
}

// ---------------------------------------------------------------------------
// RoPE (llama rotate-half) on fp32 GEMM output, emit bf16.
// Input/output [ROWS, D_] where col = h*128 + d, position s = row % S_.
// ---------------------------------------------------------------------------
__global__ __launch_bounds__(256)
void rope_cvt_kernel(const float* __restrict__ x, bf16* __restrict__ o)
{
  const unsigned idx = blockIdx.x * 256u + threadIdx.x;   // over ROWS*D_
  const int col = (int)(idx & (D_ - 1));
  const int row = (int)(idx >> 11);
  const int s   = row & (S_ - 1);
  const int d   = col & (HD_ - 1);
  const int j   = d & 63;
  const float freq = __powf(10000.0f, -(float)j * (1.0f / 64.0f));
  float sn, cs;
  __sincosf((float)s * freq, &sn, &cs);
  const float xv = x[idx];
  const float other = (d < 64) ? -x[idx + 64] : x[idx - 64];
  o[idx] = (bf16)(xv * cs + other * sn);
}

__global__ __launch_bounds__(256)
void cvt_bf16_kernel(const float* __restrict__ x, bf16* __restrict__ o, int n)
{
  const int idx = blockIdx.x * 256 + threadIdx.x;
  if (idx < n) o[idx] = (bf16)x[idx];
}

__global__ __launch_bounds__(256)
void add_residual_kernel(const float* __restrict__ a, const float* __restrict__ b,
                         float* __restrict__ hs2, float* __restrict__ out, int n)
{
  const int idx = blockIdx.x * 256 + threadIdx.x;
  if (idx < n) { const float v = a[idx] + b[idx]; hs2[idx] = v; out[idx] = v; }
}

__global__ __launch_bounds__(256)
void silu_mul_kernel(const float* __restrict__ g, const float* __restrict__ u,
                     bf16* __restrict__ m, int n)
{
  const int idx = blockIdx.x * 256 + threadIdx.x;
  if (idx < n) {
    const float gv = g[idx];
    const float s = 1.0f / (1.0f + __expf(-gv));
    m[idx] = (bf16)(gv * s * u[idx]);
  }
}

// ---------------------------------------------------------------------------
// Flash attention, all-WMMA. Block = one (b,h) x 64 queries; 128 threads.
// Each wave owns 16 query rows. K/V staged in LDS per 32-key tile.
// ---------------------------------------------------------------------------
__global__ __launch_bounds__(128)
void flash_attn_kernel(const bf16* __restrict__ Q, const bf16* __restrict__ K,
                       const bf16* __restrict__ V, bf16* __restrict__ O)
{
  const int bh   = blockIdx.y;
  const int b    = bh >> 4;          // H_ = 16
  const int h    = bh & 15;
  const int q0   = blockIdx.x * 64;
  const int tid  = threadIdx.x;
  const int lane = tid & 31;
  const int wave = tid >> 5;
  const int half = lane >> 4;
  const int l16  = lane & 15;
  const int kb   = half * 8;
  const int h16  = half * 16;
  const size_t rs = (size_t)H_ * HD_;     // 2048

  const bf16* Qb = Q + ((size_t)b * S_) * rs + (size_t)h * HD_;
  const bf16* Kb = K + ((size_t)b * S_) * rs + (size_t)h * HD_;
  const bf16* Vb = V + ((size_t)b * S_) * rs + (size_t)h * HD_;

  __shared__ bf16 Ks[32][HD_ + 8];        // keys x headdim
  __shared__ bf16 Vt[HD_][32 + 8];        // headdim x keys (transposed)
  __shared__ bf16 Ps[4][16][32 + 8];      // per-wave probability tiles

  // Q fragments for this wave's 16 rows (HD_=128 -> 4 chunks of K=32)
  v16bf qf[4];
  {
    const bf16* qrow = Qb + (size_t)(q0 + wave * 16 + l16) * rs;
    for (int c = 0; c < 4; ++c) qf[c] = frag_a(qrow + c * 32, kb);
  }

  float mrow[8], lrow[8];
  v8f oacc[8];
  for (int r = 0; r < 8; ++r) { mrow[r] = -3.0e38f; lrow[r] = 0.0f; }
  for (int nt = 0; nt < 8; ++nt)
    for (int r = 0; r < 8; ++r) oacc[nt][r] = 0.0f;

  const float scale = 0.08838834764831845f;   // 1/sqrt(128)
  const int ktend = q0 + 64;                  // causal tile bound (uniform per block)

  for (int kt = 0; kt < ktend; kt += 32) {
    __syncthreads();
    {
      const int r  = tid >> 2;               // 0..31 key rows
      const int cg = (tid & 3) * 32;         // 4 column groups of 32
      const uint4* src = (const uint4*)(Kb + (size_t)(kt + r) * rs + cg);
      uint4* dst = (uint4*)&Ks[r][cg];
      dst[0] = src[0]; dst[1] = src[1]; dst[2] = src[2]; dst[3] = src[3];
      const bf16* vsrc = Vb + (size_t)(kt + r) * rs + cg;
      for (int j = 0; j < 32; ++j) Vt[cg + j][r] = vsrc[j];
    }
    __syncthreads();

    // S tile = Q(16x128) @ K^T(128x32): 2 n-subtiles, 4 K-chunks
    v8f sacc[2];
    for (int j = 0; j < 2; ++j)
      for (int r = 0; r < 8; ++r) sacc[j][r] = 0.0f;
    for (int c = 0; c < 4; ++c) {
      const v16bf b0 = frag_b(&Ks[l16][c * 32], h16);
      const v16bf b1 = frag_b(&Ks[16 + l16][c * 32], h16);
      sacc[0] = wmma_bf16(qf[c], b0, sacc[0]);
      sacc[1] = wmma_bf16(qf[c], b1, sacc[1]);
    }

    // Scale + causal mask + per-row max (rows r+8*half per lane half)
    float s0a[8], s1a[8], rmax[8];
    for (int r = 0; r < 8; ++r) {
      const int qrow = q0 + wave * 16 + r + 8 * half;
      float s0 = sacc[0][r] * scale;
      float s1 = sacc[1][r] * scale;
      if (kt + l16 > qrow)      s0 = -1.0e30f;
      if (kt + 16 + l16 > qrow) s1 = -1.0e30f;
      s0a[r] = s0; s1a[r] = s1;
      rmax[r] = fmaxf(s0, s1);
    }
    for (int o = 1; o < 16; o <<= 1)
      for (int r = 0; r < 8; ++r)
        rmax[r] = fmaxf(rmax[r], __shfl_xor(rmax[r], o, 32));

    // Online softmax update + write P (bf16) to LDS for re-layout
    float psum[8];
    for (int r = 0; r < 8; ++r) {
      const float mnew = fmaxf(mrow[r], rmax[r]);
      const float corr = __expf(mrow[r] - mnew);
      const float p0 = __expf(s0a[r] - mnew);
      const float p1 = __expf(s1a[r] - mnew);
      mrow[r] = mnew;
      psum[r] = p0 + p1;
      lrow[r] *= corr;
      for (int nt = 0; nt < 8; ++nt) oacc[nt][r] *= corr;
      Ps[wave][r + 8 * half][l16]      = (bf16)p0;
      Ps[wave][r + 8 * half][16 + l16] = (bf16)p1;
    }
    for (int o = 1; o < 16; o <<= 1)
      for (int r = 0; r < 8; ++r)
        psum[r] += __shfl_xor(psum[r], o, 32);
    for (int r = 0; r < 8; ++r) lrow[r] += psum[r];

    __syncthreads();   // make Ps visible cross-lane (uniform barrier)

    // O += P(16x32) @ V(32x128): 8 hd-subtiles
    const v16bf pf = frag_a(&Ps[wave][l16][0], kb);
    for (int nt = 0; nt < 8; ++nt) {
      const v16bf vf = frag_b(&Vt[nt * 16 + l16][0], h16);
      oacc[nt] = wmma_bf16(pf, vf, oacc[nt]);
    }
  }

  // Normalize and store ctx bf16 [B,S,H*HD]
  for (int nt = 0; nt < 8; ++nt) {
    for (int r = 0; r < 8; ++r) {
      const int qrow = q0 + wave * 16 + r + 8 * half;
      const float invl = 1.0f / lrow[r];
      O[((size_t)b * S_ + qrow) * rs + (size_t)h * HD_ + nt * 16 + l16] =
          (bf16)(oacc[nt][r] * invl);
    }
  }
}

// ---------------------------------------------------------------------------
// Host orchestration
// ---------------------------------------------------------------------------
extern "C" void kernel_launch(void* const* d_in, const int* in_sizes, int n_in,
                              void* d_out, int out_size, void* d_ws, size_t ws_size,
                              hipStream_t stream)
{
  (void)in_sizes; (void)n_in; (void)out_size; (void)ws_size;

  const float* hidden        = (const float*)d_in[0];
  const unsigned char* mask  = (const unsigned char*)d_in[1];
  const float* wq   = (const float*)d_in[2];
  const float* wk   = (const float*)d_in[3];
  const float* wv   = (const float*)d_in[4];
  const float* wo   = (const float*)d_in[5];
  const float* ln_i = (const float*)d_in[6];
  const float* ln_p = (const float*)d_in[7];
  const float* ln_a = (const float*)d_in[8];
  const float* wgt  = (const float*)d_in[9];
  const float* wut  = (const float*)d_in[10];
  const float* wdt  = (const float*)d_in[11];
  const float* wga  = (const float*)d_in[12];
  const float* wua  = (const float*)d_in[13];
  const float* wda  = (const float*)d_in[14];
  float* out = (float*)d_out;

  char* ws = (char*)d_ws;
  size_t off = 0;
  auto take = [&](size_t bytes) -> char* {
    char* p = ws + off;
    off += (bytes + 255) & ~(size_t)255;
    return p;
  };

  // bf16 transposed weights
  bf16* wqT = (bf16*)take((size_t)D_ * D_ * 2);
  bf16* wkT = (bf16*)take((size_t)D_ * D_ * 2);
  bf16* wvT = (bf16*)take((size_t)D_ * D_ * 2);
  bf16* woT = (bf16*)take((size_t)D_ * D_ * 2);
  bf16* wgtT = (bf16*)take((size_t)D_ * F_ * 2);
  bf16* wutT = (bf16*)take((size_t)D_ * F_ * 2);
  bf16* wdtT = (bf16*)take((size_t)D_ * F_ * 2);
  bf16* wgaT = (bf16*)take((size_t)D_ * F_ * 2);
  bf16* wuaT = (bf16*)take((size_t)D_ * F_ * 2);
  bf16* wdaT = (bf16*)take((size_t)D_ * F_ * 2);
  // activations
  bf16* hln  = (bf16*)take((size_t)ROWS * D_ * 2);
  bf16* qb   = (bf16*)take((size_t)ROWS * D_ * 2);
  bf16* kbuf = (bf16*)take((size_t)ROWS * D_ * 2);
  bf16* vb   = (bf16*)take((size_t)ROWS * D_ * 2);
  bf16* ctx  = (bf16*)take((size_t)ROWS * D_ * 2);
  float* hs2 = (float*)take((size_t)ROWS * D_ * 4);
  float* t1  = (float*)take((size_t)ROWS * F_ * 4);
  float* t2  = (float*)take((size_t)ROWS * F_ * 4);
  bf16* mbuf = (bf16*)take((size_t)ROWS * F_ * 2);

  const int nD = ROWS * D_;   // 8,388,608
  const int nF = ROWS * F_;   // 33,554,432
  const dim3 blk256(256), blk128(128);

  // 1) weight prep (transpose + fp32->bf16)
  transpose_cvt_kernel<<<dim3(D_/32, D_/32), blk256, 0, stream>>>(wq, wqT, D_, D_);
  transpose_cvt_kernel<<<dim3(D_/32, D_/32), blk256, 0, stream>>>(wk, wkT, D_, D_);
  transpose_cvt_kernel<<<dim3(D_/32, D_/32), blk256, 0, stream>>>(wv, wvT, D_, D_);
  transpose_cvt_kernel<<<dim3(D_/32, D_/32), blk256, 0, stream>>>(wo, woT, D_, D_);
  transpose_cvt_kernel<<<dim3(F_/32, D_/32), blk256, 0, stream>>>(wgt, wgtT, D_, F_);
  transpose_cvt_kernel<<<dim3(F_/32, D_/32), blk256, 0, stream>>>(wut, wutT, D_, F_);
  transpose_cvt_kernel<<<dim3(D_/32, F_/32), blk256, 0, stream>>>(wdt, wdtT, F_, D_);
  transpose_cvt_kernel<<<dim3(F_/32, D_/32), blk256, 0, stream>>>(wga, wgaT, D_, F_);
  transpose_cvt_kernel<<<dim3(F_/32, D_/32), blk256, 0, stream>>>(wua, wuaT, D_, F_);
  transpose_cvt_kernel<<<dim3(D_/32, F_/32), blk256, 0, stream>>>(wda, wdaT, F_, D_);

  // 2) input RMSNorm -> bf16
  rmsnorm_kernel<<<dim3(ROWS), blk256, 0, stream>>>(hidden, ln_i, hln);

  // 3) Q/K/V projections (WMMA GEMM) + RoPE / convert
  gemm_bf16_wmma<<<dim3(D_/128, ROWS/128), blk256, 0, stream>>>(
      hln, wqT, t1, ROWS, D_, D_, 0, nullptr, 0);
  rope_cvt_kernel<<<dim3(nD/256), blk256, 0, stream>>>(t1, qb);
  gemm_bf16_wmma<<<dim3(D_/128, ROWS/128), blk256, 0, stream>>>(
      hln, wkT, t1, ROWS, D_, D_, 0, nullptr, 0);
  rope_cvt_kernel<<<dim3(nD/256), blk256, 0, stream>>>(t1, kbuf);
  gemm_bf16_wmma<<<dim3(D_/128, ROWS/128), blk256, 0, stream>>>(
      hln, wvT, t1, ROWS, D_, D_, 0, nullptr, 0);
  cvt_bf16_kernel<<<dim3(nD/256), blk256, 0, stream>>>(t1, vb, nD);

  // 4) flash attention (WMMA)
  flash_attn_kernel<<<dim3(S_/64, B_*H_), blk128, 0, stream>>>(qb, kbuf, vb, ctx);

  // 5) output projection + residual
  gemm_bf16_wmma<<<dim3(D_/128, ROWS/128), blk256, 0, stream>>>(
      ctx, woT, t1, ROWS, D_, D_, 0, nullptr, 0);
  add_residual_kernel<<<dim3(nD/256), blk256, 0, stream>>>(hidden, t1, hs2, out, nD);

  // 6) text FFN -> masked accumulate into out (rows where mask==0)
  rmsnorm_kernel<<<dim3(ROWS), blk256, 0, stream>>>(hs2, ln_p, hln);
  gemm_bf16_wmma<<<dim3(F_/128, ROWS/128), blk256, 0, stream>>>(
      hln, wgtT, t1, ROWS, F_, D_, 0, nullptr, 0);
  gemm_bf16_wmma<<<dim3(F_/128, ROWS/128), blk256, 0, stream>>>(
      hln, wutT, t2, ROWS, F_, D_, 0, nullptr, 0);
  silu_mul_kernel<<<dim3(nF/256), blk256, 0, stream>>>(t1, t2, mbuf, nF);
  gemm_bf16_wmma<<<dim3(D_/128, ROWS/128), blk256, 0, stream>>>(
      mbuf, wdtT, out, ROWS, D_, F_, 1, mask, 0);

  // 7) audio FFN -> masked accumulate into out (rows where mask!=0)
  rmsnorm_kernel<<<dim3(ROWS), blk256, 0, stream>>>(hs2, ln_a, hln);
  gemm_bf16_wmma<<<dim3(F_/128, ROWS/128), blk256, 0, stream>>>(
      hln, wgaT, t1, ROWS, F_, D_, 0, nullptr, 0);
  gemm_bf16_wmma<<<dim3(F_/128, ROWS/128), blk256, 0, stream>>>(
      hln, wuaT, t2, ROWS, F_, D_, 0, nullptr, 0);
  silu_mul_kernel<<<dim3(nF/256), blk256, 0, stream>>>(t1, t2, mbuf, nF);
  gemm_bf16_wmma<<<dim3(D_/128, ROWS/128), blk256, 0, stream>>>(
      mbuf, wdaT, out, ROWS, D_, F_, 1, mask, 1);
}
